// MultiheadAttention_10763188043786
// MI455X (gfx1250) — matrix operand order repaired and checked by reading, not compiled, not run on hardware
//
#include <hip/hip_runtime.h>

typedef __attribute__((ext_vector_type(16))) _Float16 v16h;
typedef __attribute__((ext_vector_type(8)))  _Float16 v8h;
typedef __attribute__((ext_vector_type(8)))  float    v8f;
typedef __attribute__((ext_vector_type(4)))  unsigned int u32x4;
typedef __attribute__((ext_vector_type(8)))  int      i32x8;
typedef __attribute__((ext_vector_type(4)))  int      i32x4;

#define E_   768
#define H_   12
#define DK_  64
#define B_   4
#define S_   2048
#define MTOT (B_ * S_)   // 8192

// ---------------------------------------------------------------- helpers
static __device__ inline v8f zero_v8f() {
  v8f z;
#pragma unroll
  for (int i = 0; i < 8; ++i) z[i] = 0.0f;
  return z;
}

static __device__ inline v16h pack16(v8h lo, v8h hi) {
  v16h r;
#pragma unroll
  for (int j = 0; j < 8; ++j) { r[j] = lo[j]; r[8 + j] = hi[j]; }
  return r;
}

// TDM: async 2D f16 tile load global->LDS (D# per CDNA5 ISA).
// tile_d0 = contiguous elements per row, tile_d1 = rows, stride = row pitch (elems).
// 6-arg builtin form (clang-23 / therock-10.0 lane).
static __device__ inline void tdm_load_2d_f16(unsigned lds_off, const _Float16* gptr,
                                              unsigned tile_d0, unsigned tile_d1,
                                              unsigned tensor_d0, unsigned tensor_d1,
                                              unsigned stride_elems) {
  const unsigned long long ga = (unsigned long long)(uintptr_t)gptr;
  u32x4 g0;
  g0[0] = 1u;                                        // count=1, user descriptor
  g0[1] = lds_off;                                   // lds_addr (bytes)
  g0[2] = (unsigned)(ga & 0xffffffffu);              // global_addr[31:0]
  g0[3] = (unsigned)((ga >> 32) & 0x01ffffffu) | (2u << 30);  // addr[56:32] | type=2
  i32x8 g1;
  g1[0] = (int)(1u << 16);                           // wg_mask=0, data_size=1 (2B)
  g1[1] = (int)((tensor_d0 & 0xffffu) << 16);        // tensor_dim0[15:0]
  g1[2] = (int)(((tensor_d0 >> 16) & 0xffffu) | ((tensor_d1 & 0xffffu) << 16));
  g1[3] = (int)(((tensor_d1 >> 16) & 0xffffu) | ((tile_d0 & 0xffffu) << 16));
  g1[4] = (int)(tile_d1 & 0xffffu);                  // tile_dim1, tile_dim2=0
  g1[5] = (int)stride_elems;                         // tensor_dim0_stride[31:0]
  g1[6] = 0;
  g1[7] = 0;
  i32x4 z4;
  z4[0] = 0; z4[1] = 0; z4[2] = 0; z4[3] = 0;        // 2D: groups 2/3 unused
  i32x8 z8;
#pragma unroll
  for (int j = 0; j < 8; ++j) z8[j] = 0;
  __builtin_amdgcn_tensor_load_to_lds(g0, g1, z4, z4, z8, 0);
}

// ============================================================== kernel 0
// Elementwise fp32 -> f16 conversion (activations and weights), done once so
// the GEMM inner loops are pure f16 loads + WMMA.
__global__ __launch_bounds__(256) void cvt16_kernel(const float* __restrict__ src,
                                                    _Float16* __restrict__ dst,
                                                    int n8) {
  const int i = blockIdx.x * 256 + threadIdx.x;
  if (i < n8) {
    v8f v = *(const v8f*)(src + (size_t)i * 8);
    v8h o;
#pragma unroll
    for (int j = 0; j < 8; ++j) o[j] = (_Float16)v[j];
    *(v8h*)(dst + (size_t)i * 8) = o;
  }
}

// ============================================================== kernel 1
// Q/K/V projections from pre-converted f16: one wave = 64x32 output tile,
// 8 v_wmma per K-step.  Q,K -> [B,H,S,DK]; V -> transposed [B,H,DK,S].
__global__ __launch_bounds__(32) void qkv_proj_kernel(
    const _Float16* __restrict__ Xq, const _Float16* __restrict__ Xk,
    const _Float16* __restrict__ Xv,
    const _Float16* __restrict__ Wq, const _Float16* __restrict__ Wk,
    const _Float16* __restrict__ Wv,
    _Float16* __restrict__ Qp, _Float16* __restrict__ Kp,
    _Float16* __restrict__ Vt)
{
  const int lane  = threadIdx.x & 31;
  const int lhalf = lane >> 4;
  const int lmod  = lane & 15;
  const int nbase = blockIdx.x * 32;
  const int mbase = blockIdx.y * 64;
  const int which = blockIdx.z;

  const _Float16* X = (which == 0) ? Xq : (which == 1) ? Xk : Xv;
  const _Float16* W = (which == 0) ? Wq : (which == 1) ? Wk : Wv;

  v8f acc[4][2];
#pragma unroll
  for (int mi = 0; mi < 4; ++mi)
#pragma unroll
    for (int ni = 0; ni < 2; ++ni) acc[mi][ni] = zero_v8f();

  for (int kk = 0; kk < E_; kk += 32) {
    v16h a[4], bt[2];
#pragma unroll
    for (int mi = 0; mi < 4; ++mi) {
      // A layout: lanes<16 K={0..7,16..23}, lanes>=16 K={8..15,24..31}
      const _Float16* ap = X + (size_t)(mbase + mi * 16 + lmod) * E_ + kk + (lhalf ? 8 : 0);
      a[mi] = pack16(*(const v8h*)ap, *(const v8h*)(ap + 16));
    }
#pragma unroll
    for (int ni = 0; ni < 2; ++ni) {
      // B layout: lane = column n (row of W), lanes<16 K={0..7,16..23}, lanes>=16 K={8..15,24..31}
      const _Float16* bp = W + (size_t)(nbase + ni * 16 + lmod) * E_ + kk + (lhalf ? 8 : 0);
      bt[ni] = pack16(*(const v8h*)bp, *(const v8h*)(bp + 16));
    }
#pragma unroll
    for (int mi = 0; mi < 4; ++mi)
#pragma unroll
      for (int ni = 0; ni < 2; ++ni)
        acc[mi][ni] = __builtin_amdgcn_wmma_f32_16x16x32_f16(
            false, a[mi], false, bt[ni], (short)0, acc[mi][ni], false, false);
  }

#pragma unroll
  for (int mi = 0; mi < 4; ++mi)
#pragma unroll
    for (int ni = 0; ni < 2; ++ni)
#pragma unroll
      for (int r = 0; r < 8; ++r) {
        const int m = mbase + mi * 16 + r + lhalf * 8;
        const int n = nbase + ni * 16 + lmod;
        const int bb = m >> 11;
        const int ss = m & (S_ - 1);
        const int hh = n / DK_;
        const int dd = n & (DK_ - 1);
        const _Float16 val = (_Float16)acc[mi][ni][r];
        if (which == 2)
          Vt[(((size_t)bb * H_ + hh) * DK_ + dd) * S_ + ss] = val;
        else if (which == 0)
          Qp[(((size_t)bb * H_ + hh) * S_ + ss) * DK_ + dd] = val;
        else
          Kp[(((size_t)bb * H_ + hh) * S_ + ss) * DK_ + dd] = val;
      }
}

// ============================================================== kernel 2
// Flash-style attention with TDM double-buffered K/V staging into LDS.
__global__ __launch_bounds__(32) void attn_kernel(
    const _Float16* __restrict__ Qp, const _Float16* __restrict__ Kp,
    const _Float16* __restrict__ Vt, const unsigned char* __restrict__ mask,
    _Float16* __restrict__ Ctx)
{
  __shared__ __align__(16) _Float16 sK[2][32 * DK_];  // 32 keys x 64 d (row-major d)
  __shared__ __align__(16) _Float16 sV[2][DK_ * 32];  // 64 d x 32 keys (row-major keys)
  __shared__ __align__(32) float    sS[16 * 32];      // raw score tile
  __shared__ __align__(32) _Float16 sP[16 * 40];      // exp(P), padded rows (80B)
  __shared__ __align__(32) float    sScale[16];
  __shared__ __align__(32) float    sSum[16];

  const int lane  = threadIdx.x & 31;
  const int lhalf = lane >> 4;
  const int lmod  = lane & 15;
  const int qbase = blockIdx.x * 16;
  const int bh    = blockIdx.y;
  const int bidx  = bh / H_;

  const _Float16* Qh = Qp + (size_t)bh * S_ * DK_;
  const _Float16* Kh = Kp + (size_t)bh * S_ * DK_;
  const _Float16* Vh = Vt + (size_t)bh * DK_ * S_;
  const unsigned char* Mb = mask + (size_t)bidx * S_ * S_;

  const unsigned kOff0 = (unsigned)(uintptr_t)&sK[0][0];
  const unsigned kOff1 = (unsigned)(uintptr_t)&sK[1][0];
  const unsigned vOff0 = (unsigned)(uintptr_t)&sV[0][0];
  const unsigned vOff1 = (unsigned)(uintptr_t)&sV[1][0];

  // Preload Q A-tiles (K-chunks d=0..31, d=32..63)
  v16h aQ[2];
#pragma unroll
  for (int c = 0; c < 2; ++c) {
    const _Float16* qp = Qh + (size_t)(qbase + lmod) * DK_ + c * 32 + (lhalf ? 8 : 0);
    aQ[c] = pack16(*(const v8h*)qp, *(const v8h*)(qp + 16));
  }

  // TDM prime: stage first K/V tiles into buffer 0
  tdm_load_2d_f16(kOff0, Kh, DK_, 32, DK_, S_, DK_);
  tdm_load_2d_f16(vOff0, Vh, 32, DK_, S_, DK_, S_);

  v8f ctx[4];
#pragma unroll
  for (int dt = 0; dt < 4; ++dt) ctx[dt] = zero_v8f();
  float rowm = -__builtin_inff();
  float rowl = 0.0f;

  for (int kb = 0; kb < S_; kb += 32) {
    const int cur = (kb >> 5) & 1;

    // Issue next tile into the other buffer, then wait for current buffer.
    asm volatile("s_wait_dscnt 0x0" ::: "memory");
    if (kb + 32 < S_) {
      const unsigned kn = cur ? kOff0 : kOff1;
      const unsigned vn = cur ? vOff0 : vOff1;
      tdm_load_2d_f16(kn, Kh + (size_t)(kb + 32) * DK_, DK_, 32, DK_, S_, DK_);
      tdm_load_2d_f16(vn, Vh + kb + 32, 32, DK_, S_, DK_, S_);
      asm volatile("s_wait_tensorcnt 0x2" ::: "memory");
    } else {
      asm volatile("s_wait_tensorcnt 0x0" ::: "memory");
    }

    // ---- scores S = Q @ K^T from staged LDS K tile
    v8f sc[2];
#pragma unroll
    for (int nt = 0; nt < 2; ++nt) {
      v8f z = zero_v8f();
#pragma unroll
      for (int c = 0; c < 2; ++c) {
        const _Float16* kp = &sK[cur][(nt * 16 + lmod) * DK_ + c * 32 + (lhalf ? 8 : 0)];
        v16h bk = pack16(*(const v8h*)kp, *(const v8h*)(kp + 16));
        z = __builtin_amdgcn_wmma_f32_16x16x32_f16(
            false, aQ[c], false, bk, (short)0, z, false, false);
      }
      sc[nt] = z;
    }

    // ---- stage raw scores to LDS (C layout -> row-major)
#pragma unroll
    for (int nt = 0; nt < 2; ++nt)
#pragma unroll
      for (int r = 0; r < 8; ++r)
        sS[(r + lhalf * 8) * 32 + nt * 16 + lmod] = sc[nt][r];
    asm volatile("s_wait_dscnt 0x0" ::: "memory");

    // ---- online softmax + mask: lanes 0..15 each own one query row
    if (lane < 16) {
      const unsigned char* mr = Mb + (size_t)(qbase + lane) * S_ + kb;
      const uint4 mm0 = *(const uint4*)(mr);
      const uint4 mm1 = *(const uint4*)(mr + 16);
      unsigned mw[8];
      mw[0] = mm0.x; mw[1] = mm0.y; mw[2] = mm0.z; mw[3] = mm0.w;
      mw[4] = mm1.x; mw[5] = mm1.y; mw[6] = mm1.z; mw[7] = mm1.w;
      float sv[32];
      float mx = rowm;
#pragma unroll
      for (int j = 0; j < 32; ++j) {
        float s = sS[lane * 32 + j] * 0.125f;            // 1/sqrt(64)
        if ((mw[j >> 2] >> (8 * (j & 3))) & 0xff) s = -1.0e9f;
        sv[j] = s;
        mx = fmaxf(mx, s);
      }
      float sum = 0.0f;
#pragma unroll
      for (int j = 0; j < 32; ++j) {
        const float p = __expf(sv[j] - mx);
        sum += p;
        sP[lane * 40 + j] = (_Float16)p;                 // padded stride: no conflicts
      }
      const float corr = __expf(rowm - mx);
      rowl = rowl * corr + sum;
      rowm = mx;
      sScale[lane] = corr;
    }
    asm volatile("s_wait_dscnt 0x0" ::: "memory");

    // ---- reload P in WMMA A layout, rescale accumulators, P @ V from LDS V
    const _Float16* pp = &sP[lmod * 40 + (lhalf ? 8 : 0)];
    v16h aP = pack16(*(const v8h*)pp, *(const v8h*)(pp + 16));
    v8f corrv = *(const v8f*)(&sScale[lhalf * 8]);

#pragma unroll
    for (int dt = 0; dt < 4; ++dt) {
#pragma unroll
      for (int r = 0; r < 8; ++r) ctx[dt][r] *= corrv[r];
      const _Float16* vp = &sV[cur][(dt * 16 + lmod) * 32 + (lhalf ? 8 : 0)];
      v16h bv = pack16(*(const v8h*)vp, *(const v8h*)(vp + 16));
      ctx[dt] = __builtin_amdgcn_wmma_f32_16x16x32_f16(
          false, aP, false, bv, (short)0, ctx[dt], false, false);
    }
  }

  // ---- finalize: divide by row sums, store context f16 [B,H,S,DK]
  if (lane < 16) sSum[lane] = rowl;
  asm volatile("s_wait_dscnt 0x0" ::: "memory");
  v8f sumv = *(const v8f*)(&sSum[lhalf * 8]);
#pragma unroll
  for (int dt = 0; dt < 4; ++dt)
#pragma unroll
    for (int r = 0; r < 8; ++r) {
      const int qr = qbase + r + lhalf * 8;
      const float o = ctx[dt][r] / sumv[r];
      Ctx[((size_t)bh * S_ + qr) * DK_ + dt * 16 + lmod] = (_Float16)o;
    }
}

// ============================================================== kernel 3
// out_pre = context @ Wo^T + query (residual).  64x32 tile per wave.
__global__ __launch_bounds__(32) void out_proj_kernel(
    const _Float16* __restrict__ Ctx, const _Float16* __restrict__ Wo,
    const float* __restrict__ query, float* __restrict__ outpre)
{
  const int lane  = threadIdx.x & 31;
  const int lhalf = lane >> 4;
  const int lmod  = lane & 15;
  const int nbase = blockIdx.x * 32;
  const int mbase = blockIdx.y * 64;

  v8f acc[4][2];
#pragma unroll
  for (int mi = 0; mi < 4; ++mi)
#pragma unroll
    for (int ni = 0; ni < 2; ++ni) acc[mi][ni] = zero_v8f();

  for (int kk = 0; kk < E_; kk += 32) {
    v16h a[4], bt[2];
#pragma unroll
    for (int mi = 0; mi < 4; ++mi) {
      const int m  = mbase + mi * 16 + lmod;
      const int bb = m >> 11;
      const int ss = m & (S_ - 1);
      const int k0 = kk + (lhalf ? 8 : 0);
      const int k1 = k0 + 16;
      const _Float16* p0 =
          Ctx + (((size_t)bb * H_ + k0 / DK_) * S_ + ss) * DK_ + (k0 & (DK_ - 1));
      const _Float16* p1 =
          Ctx + (((size_t)bb * H_ + k1 / DK_) * S_ + ss) * DK_ + (k1 & (DK_ - 1));
      a[mi] = pack16(*(const v8h*)p0, *(const v8h*)p1);
    }
#pragma unroll
    for (int ni = 0; ni < 2; ++ni) {
      const _Float16* bp = Wo + (size_t)(nbase + ni * 16 + lmod) * E_ + kk + (lhalf ? 8 : 0);
      bt[ni] = pack16(*(const v8h*)bp, *(const v8h*)(bp + 16));
    }
#pragma unroll
    for (int mi = 0; mi < 4; ++mi)
#pragma unroll
      for (int ni = 0; ni < 2; ++ni)
        acc[mi][ni] = __builtin_amdgcn_wmma_f32_16x16x32_f16(
            false, a[mi], false, bt[ni], (short)0, acc[mi][ni], false, false);
  }

#pragma unroll
  for (int mi = 0; mi < 4; ++mi)
#pragma unroll
    for (int ni = 0; ni < 2; ++ni)
#pragma unroll
      for (int r = 0; r < 8; ++r) {
        const int m = mbase + mi * 16 + r + lhalf * 8;
        const int n = nbase + ni * 16 + lmod;
        outpre[(size_t)m * E_ + n] = acc[mi][ni][r] + query[(size_t)m * E_ + n];
      }
}

// ============================================================== kernel 4
// LayerNorm over E=768 per row (weight=1, bias=0, eps=1e-5).
__global__ __launch_bounds__(256) void ln_kernel(const float* __restrict__ x,
                                                 float* __restrict__ out)
{
  __shared__ float s1[256];
  __shared__ float s2[256];
  const int row = blockIdx.x;
  const int tid = threadIdx.x;
  const float* xr = x + (size_t)row * E_;
  const float v0 = xr[tid], v1 = xr[tid + 256], v2 = xr[tid + 512];
  s1[tid] = v0 + v1 + v2;
  s2[tid] = v0 * v0 + v1 * v1 + v2 * v2;
  __syncthreads();
  for (int off = 128; off > 0; off >>= 1) {
    if (tid < off) { s1[tid] += s1[tid + off]; s2[tid] += s2[tid + off]; }
    __syncthreads();
  }
  const float mean = s1[0] * (1.0f / E_);
  const float var  = s2[0] * (1.0f / E_) - mean * mean;
  const float inv  = rsqrtf(var + 1.0e-5f);
  float* orow = out + (size_t)row * E_;
  orow[tid]       = (v0 - mean) * inv;
  orow[tid + 256] = (v1 - mean) * inv;
  orow[tid + 512] = (v2 - mean) * inv;
}

// ============================================================== launcher
extern "C" void kernel_launch(void* const* d_in, const int* in_sizes, int n_in,
                              void* d_out, int out_size, void* d_ws, size_t ws_size,
                              hipStream_t stream) {
  (void)in_sizes; (void)n_in; (void)out_size; (void)ws_size;
  const float* q  = (const float*)d_in[0];
  const float* k  = (const float*)d_in[1];
  const float* v  = (const float*)d_in[2];
  const unsigned char* mask = (const unsigned char*)d_in[3];
  const float* Wq = (const float*)d_in[4];
  const float* Wk = (const float*)d_in[5];
  const float* Wv = (const float*)d_in[6];
  const float* Wo = (const float*)d_in[7];
  float* out = (float*)d_out;

  const size_t nAct = (size_t)MTOT * E_;       // 6,291,456
  const size_t nW   = (size_t)E_ * E_;         // 589,824
  char* ws = (char*)d_ws;
  _Float16* Xq16 = (_Float16*)ws;
  _Float16* Xk16 = Xq16 + nAct;
  _Float16* Xv16 = Xk16 + nAct;
  _Float16* Wq16 = Xv16 + nAct;
  _Float16* Wk16 = Wq16 + nW;
  _Float16* Wv16 = Wk16 + nW;
  _Float16* Wo16 = Wv16 + nW;
  _Float16* Qp   = Wo16 + nW;
  _Float16* Kp   = Qp + nAct;
  _Float16* Vt   = Kp + nAct;
  _Float16* Ctx  = Vt + nAct;
  float*    outpre = (float*)(Ctx + nAct);

  // --- fp32 -> f16 conversion passes
  const int actB = (int)((nAct / 8 + 255) / 256);
  const int wB   = (int)((nW   / 8 + 255) / 256);
  cvt16_kernel<<<actB, 256, 0, stream>>>(q,  Xq16, (int)(nAct / 8));
  cvt16_kernel<<<actB, 256, 0, stream>>>(k,  Xk16, (int)(nAct / 8));
  cvt16_kernel<<<actB, 256, 0, stream>>>(v,  Xv16, (int)(nAct / 8));
  cvt16_kernel<<<wB,   256, 0, stream>>>(Wq, Wq16, (int)(nW / 8));
  cvt16_kernel<<<wB,   256, 0, stream>>>(Wk, Wk16, (int)(nW / 8));
  cvt16_kernel<<<wB,   256, 0, stream>>>(Wv, Wv16, (int)(nW / 8));
  cvt16_kernel<<<wB,   256, 0, stream>>>(Wo, Wo16, (int)(nW / 8));

  dim3 g1(E_ / 32, MTOT / 64, 3);
  qkv_proj_kernel<<<g1, 32, 0, stream>>>(Xq16, Xk16, Xv16, Wq16, Wk16, Wv16,
                                         Qp, Kp, Vt);

  dim3 g2(S_ / 16, B_ * H_);
  attn_kernel<<<g2, 32, 0, stream>>>(Qp, Kp, Vt, mask, Ctx);

  dim3 g3(E_ / 32, MTOT / 64);
  out_proj_kernel<<<g3, 32, 0, stream>>>(Ctx, Wo16, q, outpre);

  ln_kernel<<<MTOT, 256, 0, stream>>>(outpre, out);
}
